// Model_41231686042069
// MI455X (gfx1250) — matrix operand order repaired and checked by reading, not compiled, not run on hardware
//
#include <hip/hip_runtime.h>
#include <stdint.h>
#include <stddef.h>

// ---------------- problem constants (match reference) ----------------
#define B_    64      // batch
#define T_    336     // encoder sequence length
#define TP_   96      // decoder pred_len (reference constant)
#define H_    512     // hidden
#define H4_   2048    // 4*hidden (gates)
#define EIN_  321     // enc_in
#define KIN_  352     // enc_in padded to multiple of 32 (for K of WMMA)
#define FCR_  336     // fc output rows padded to multiple of 16

typedef __bf16 bf16;
typedef __attribute__((ext_vector_type(16))) bf16  v16bf;
typedef __attribute__((ext_vector_type(8)))  bf16  v8bf;
typedef __attribute__((ext_vector_type(8)))  float v8f;

// ---------------- WMMA helpers ----------------
__device__ inline v8f wmma_bf16(v16bf a, v16bf b, v8f c) {
  // (neg_a, A, neg_b, B, c_mod, C, reuse_a, reuse_b)
  return __builtin_amdgcn_wmma_f32_16x16x32_bf16(false, a, false, b, (short)0, c,
                                                 false, false);
}

// A fragment: 16x32 bf16 tile, row-major source [rows][lda].
// Lane L (<16): row = L, holds K = {0..7} and {16..23}; lanes 16..31: K={8..15},{24..31}.
// Caller passes pointer already offset by: row*lda + ((lane>>4)<<3)  (+ kBase).
__device__ inline v16bf load_a_frag(const bf16* p) {
  v8bf lo = *(const v8bf*)(p);
  v8bf hi = *(const v8bf*)(p + 16);
  v16bf r;
#pragma unroll
  for (int i = 0; i < 8; ++i) { r[i] = lo[i]; r[i + 8] = hi[i]; }
  return r;
}

// B fragment: 32x16 bf16 tile. B[k][n] = W[n][k] with W row-major [N][ldw].
// Lane n (<16): column n, K = {0..7} and {16..23}; lanes 16..31: K = {8..15},{24..31}.
// Caller passes pointer already offset by: n*ldw + ((lane>>4)<<3)  (+ kBase).
__device__ inline v16bf load_b_frag(const bf16* p) {
  v8bf lo = *(const v8bf*)(p);
  v8bf hi = *(const v8bf*)(p + 16);
  v16bf r;
#pragma unroll
  for (int i = 0; i < 8; ++i) { r[i] = lo[i]; r[i + 8] = hi[i]; }
  return r;
}

__device__ inline const bf16* a_base(const bf16* A, int lda, int mt, int lane) {
  return A + (size_t)(mt * 16 + (lane & 15)) * lda + ((lane >> 4) << 3);
}
__device__ inline const bf16* b_base(const bf16* W, int ldw, int nt, int lane) {
  return W + (size_t)(nt * 16 + (lane & 15)) * ldw + ((lane >> 4) << 3);
}

// C/D 16x16 f32: lane<16 -> (M=i, N=lane); lane>=16 -> (M=8+i, N=lane-16)
__device__ inline void store_tile(float* C, int ldc, int mt, int nt, int lane, v8f acc) {
  float* p = C + (size_t)(mt * 16 + ((lane >> 4) << 3)) * ldc + nt * 16 + (lane & 15);
#pragma unroll
  for (int i = 0; i < 8; ++i) p[(size_t)i * ldc] = acc[i];
}
__device__ inline v8f load_tile(const float* C, int ldc, int mt, int nt, int lane) {
  const float* p = C + (size_t)(mt * 16 + ((lane >> 4) << 3)) * ldc + nt * 16 + (lane & 15);
  v8f acc;
#pragma unroll
  for (int i = 0; i < 8; ++i) acc[i] = p[(size_t)i * ldc];
  return acc;
}

// ---------------- activation helpers ----------------
__device__ inline float sigmf(float x) { return 1.0f / (1.0f + __expf(-x)); }
__device__ inline float tanhfast(float x) {
  x = fminf(fmaxf(x, -15.0f), 15.0f);
  float e = __expf(2.0f * x);
  return (e - 1.0f) / (e + 1.0f);
}

// ---------------- software grid barrier (sense via generation counter) ----------------
__device__ inline void grid_sync(unsigned* bar) {
  volatile unsigned* cnt = bar;
  volatile unsigned* gen = bar + 1;
  __syncthreads();
  if (threadIdx.x == 0) {
    __threadfence();                       // release my writes
    unsigned g = *gen;
    unsigned arrived = atomicAdd((unsigned*)cnt, 1u) + 1u;
    if (arrived == gridDim.x) {
      *cnt = 0u;
      __threadfence();
      atomicAdd((unsigned*)gen, 1u);
    } else {
      while (*gen == g) { __builtin_amdgcn_s_sleep(2); }
    }
    __threadfence();                       // acquire others' writes
  }
  __syncthreads();
}

// ---------------- LSTM elementwise cell update ----------------
// gates: [B][4H] f32 (order i,f,g,o); c: [B][H] f32; hbf: [B][H] bf16
__device__ inline void lstm_update(const float* __restrict__ gates,
                                   float* __restrict__ c,
                                   bf16* __restrict__ hbf,
                                   bf16* __restrict__ seq_out,
                                   int tid, int nthr) {
  for (int idx = tid; idx < B_ * H_; idx += nthr) {
    int b = idx >> 9;            // /512
    int j = idx & (H_ - 1);
    const float* g = gates + (size_t)b * H4_;
    float iv = sigmf(g[j]);
    float fv = sigmf(g[j + H_]);
    float gv = tanhfast(g[j + 2 * H_]);
    float ov = sigmf(g[j + 3 * H_]);
    float cn = fv * c[idx] + iv * gv;
    c[idx] = cn;
    float h = ov * tanhfast(cn);
    bf16 hb = (bf16)h;
    hbf[idx] = hb;
    if (seq_out) seq_out[idx] = hb;
  }
}

// ---------------- prep: convert/pack weights+inputs to bf16, zero state ----------------
struct PrepArgs {
  const float *x;
  const float *wih0e, *whh0e, *wih1e, *whh1e;
  const float *wih0d, *whh0d, *wih1d, *whh1d;
  const float *fcw;
  bf16 *x_bf;
  bf16 *wih0e_b, *whh0e_b, *wih1e_b, *whh1e_b;
  bf16 *wih0d_b, *whh0d_b, *wih1d_b, *whh1d_b;
  bf16 *fcw_b;
  bf16 *he0, *he1, *xt;
  float *ce0, *ce1;
  unsigned* bar;
};

__device__ inline void conv_pad(const float* src, bf16* dst, int rd, int cd,
                                int rs, int cs, int tid, int stride) {
  int total = rd * cd;
  for (int i = tid; i < total; i += stride) {
    int r = i / cd, ccol = i - r * cd;
    float v = (r < rs && ccol < cs) ? src[(size_t)r * cs + ccol] : 0.0f;
    dst[i] = (bf16)v;
  }
}

__global__ void prep_kernel(PrepArgs a) {
  int tid = blockIdx.x * blockDim.x + threadIdx.x;
  int stride = gridDim.x * blockDim.x;
  if (tid == 0) { a.bar[0] = 0u; a.bar[1] = 0u; }
  // weights (K padded to 352 for input-side matrices; fc rows padded to 336)
  conv_pad(a.wih0e, a.wih0e_b, H4_, KIN_, H4_, EIN_, tid, stride);
  conv_pad(a.whh0e, a.whh0e_b, H4_, H_,   H4_, H_,   tid, stride);
  conv_pad(a.wih1e, a.wih1e_b, H4_, H_,   H4_, H_,   tid, stride);
  conv_pad(a.whh1e, a.whh1e_b, H4_, H_,   H4_, H_,   tid, stride);
  conv_pad(a.wih0d, a.wih0d_b, H4_, KIN_, H4_, EIN_, tid, stride);
  conv_pad(a.whh0d, a.whh0d_b, H4_, H_,   H4_, H_,   tid, stride);
  conv_pad(a.wih1d, a.wih1d_b, H4_, H_,   H4_, H_,   tid, stride);
  conv_pad(a.whh1d, a.whh1d_b, H4_, H_,   H4_, H_,   tid, stride);
  conv_pad(a.fcw,   a.fcw_b,   FCR_, H_,  EIN_, H_,  tid, stride);
  // x: [B][T][321] -> bf16 [T*B][352] (row r = t*64+b), zero pad K tail
  for (int i = tid; i < T_ * B_ * KIN_; i += stride) {
    int r = i / KIN_, k = i - r * KIN_;
    int t = r >> 6, b = r & 63;
    float v = (k < EIN_) ? a.x[((size_t)b * T_ + t) * EIN_ + k] : 0.0f;
    a.x_bf[i] = (bf16)v;
  }
  // zero initial states and decoder input
  for (int i = tid; i < B_ * H_; i += stride) {
    a.he0[i] = (bf16)0.0f; a.he1[i] = (bf16)0.0f;
    a.ce0[i] = 0.0f;       a.ce1[i] = 0.0f;
  }
  for (int i = tid; i < B_ * KIN_; i += stride) a.xt[i] = (bf16)0.0f;
}

// ---------------- batched input-projection GEMM: C = A @ W^T + bias ----------------
// A: bf16 [M][lda]; W: bf16 [N][ldw]; C: f32 [M][ldc]. One 16x16 tile per wave.
// grid = (Ntiles/8, Mtiles), block = 256 (8 waves).
__global__ void __launch_bounds__(256) gemm_bias_bf16(
    const bf16* __restrict__ A, int lda,
    const bf16* __restrict__ W, int ldw,
    const float* __restrict__ bias,
    float* __restrict__ C, int ldc, int K) {
  int lane = threadIdx.x & 31;
  int wave = threadIdx.x >> 5;
  int nt = blockIdx.x * 8 + wave;
  int mt = blockIdx.y;
  const bf16* ap = a_base(A, lda, mt, lane);
  const bf16* bp = b_base(W, ldw, nt, lane);
  v8f acc = {};
  for (int k = 0; k < K; k += 32)
    acc = wmma_bf16(load_a_frag(ap + k), load_b_frag(bp + k), acc);
  float bv = bias[nt * 16 + (lane & 15)];
#pragma unroll
  for (int i = 0; i < 8; ++i) acc[i] += bv;
  store_tile(C, ldc, mt, nt, lane, acc);
}

// ---------------- persistent encoder-layer recurrence ----------------
// gates[t] = xproj[t] (incl. bias) + h @ Whh^T ; then cell update.
// grid = 64 blocks x 256 thr = 512 waves = one gate tile per wave.
// Whh B-fragments are loop-invariant: pinned in 128 VGPRs across all timesteps.
__global__ void __launch_bounds__(256) lstm_recurrence(
    const bf16* __restrict__ Whh, const float* __restrict__ xproj,
    bf16* __restrict__ hbf, float* __restrict__ cst,
    bf16* __restrict__ seq_out, float* __restrict__ gates,
    unsigned* __restrict__ bar, int T) {
  int tid  = blockIdx.x * blockDim.x + threadIdx.x;
  int nthr = gridDim.x * blockDim.x;
  int wid  = tid >> 5;
  int lane = threadIdx.x & 31;
  int mt = wid >> 7;        // 0..3   (batch tiles)
  int nt = wid & 127;       // 0..127 (gate tiles)
  const bf16* ap = a_base(hbf, H_, mt, lane);
  const bf16* bp = b_base(Whh, H_, nt, lane);
  // pin recurrent weight tile in registers (16 frags x 8 VGPRs = 128 VGPRs)
  v16bf bw[16];
#pragma unroll
  for (int k = 0; k < 16; ++k) bw[k] = load_b_frag(bp + 32 * k);

  const float* xtile0 = xproj + (size_t)(mt * 16 + ((lane >> 4) << 3)) * H4_
                       + nt * 16 + (lane & 15);
  for (int t = 0; t < T; ++t) {
    // prefetch next step's xproj C-tile (HBM stream) behind this step's WMMA
    if (t + 1 < T) {
      const float* nx = xtile0 + (size_t)(t + 1) * B_ * H4_;
#pragma unroll
      for (int i = 0; i < 8; ++i) __builtin_prefetch(nx + (size_t)i * H4_, 0, 1);
    }
    v8f acc = load_tile(xproj + (size_t)t * B_ * H4_, H4_, mt, nt, lane);
#pragma unroll
    for (int k = 0; k < 16; ++k)
      acc = wmma_bf16(load_a_frag(ap + 32 * k), bw[k], acc);
    store_tile(gates, H4_, mt, nt, lane, acc);
    grid_sync(bar);
    lstm_update(gates, cst, hbf,
                seq_out ? seq_out + (size_t)t * B_ * H_ : (bf16*)nullptr,
                tid, nthr);
    grid_sync(bar);
  }
}

// ---------------- persistent autoregressive decoder ----------------
// Pins the two recurrent weight tiles (Whh0, Whh1) in 256 VGPRs; streams
// Wih0/Wih1/fcW from L2 (whole decoder weight set ~7.6MB, L2-resident).
__global__ void __launch_bounds__(256) decoder_kernel(
    const bf16* __restrict__ Wih0, const bf16* __restrict__ Whh0, const float* __restrict__ bi0,
    const bf16* __restrict__ Wih1, const bf16* __restrict__ Whh1, const float* __restrict__ bi1,
    const bf16* __restrict__ fcW,  const float* __restrict__ fcb,
    bf16* __restrict__ xt,
    bf16* __restrict__ h0, float* __restrict__ c0,
    bf16* __restrict__ h1, float* __restrict__ c1,
    float* __restrict__ gates, float* __restrict__ out,
    unsigned* __restrict__ bar) {
  int tid  = blockIdx.x * blockDim.x + threadIdx.x;
  int nthr = gridDim.x * blockDim.x;
  int wid  = tid >> 5;
  int lane = threadIdx.x & 31;
  int mt = wid >> 7;
  int nt = wid & 127;
  const bf16* a_x  = a_base(xt, KIN_, mt, lane);
  const bf16* a_h0 = a_base(h0, H_,  mt, lane);
  const bf16* a_h1 = a_base(h1, H_,  mt, lane);
  const bf16* b_i0 = b_base(Wih0, KIN_, nt, lane);
  const bf16* b_h0 = b_base(Whh0, H_,  nt, lane);
  const bf16* b_i1 = b_base(Wih1, H_,  nt, lane);
  const bf16* b_h1 = b_base(Whh1, H_,  nt, lane);
  float bv0 = bi0[nt * 16 + (lane & 15)];
  float bv1 = bi1[nt * 16 + (lane & 15)];
  // pin recurrent weight tiles (2 x 16 frags x 8 VGPRs = 256 VGPRs)
  v16bf bwh0[16], bwh1[16];
#pragma unroll
  for (int k = 0; k < 16; ++k) {
    bwh0[k] = load_b_frag(b_h0 + 32 * k);
    bwh1[k] = load_b_frag(b_h1 + 32 * k);
  }
  // fc tiles: 4 Mtiles x 21 Ntiles (=336/16) = 84 tiles, waves 0..83
  bool do_fc = (wid < 84);
  int fmt = wid / 21, fnt = wid % 21;
  if (!do_fc) { fmt = 0; fnt = 0; }
  const bf16* a_fc = a_base(h1,  H_, fmt, lane);
  const bf16* b_fc = b_base(fcW, H_, fnt, lane);
  int fcol = fnt * 16 + (lane & 15);
  float bvf = (do_fc && fcol < EIN_) ? fcb[fcol] : 0.0f;

  for (int t = 0; t < TP_; ++t) {
    // ---- layer 0: gates = xt @ Wih0^T + h0 @ Whh0^T + b0 ----
    v8f acc;
#pragma unroll
    for (int i = 0; i < 8; ++i) acc[i] = bv0;
    for (int k = 0; k < KIN_; k += 32)
      acc = wmma_bf16(load_a_frag(a_x + k), load_b_frag(b_i0 + k), acc);
#pragma unroll
    for (int k = 0; k < 16; ++k)
      acc = wmma_bf16(load_a_frag(a_h0 + 32 * k), bwh0[k], acc);
    store_tile(gates, H4_, mt, nt, lane, acc);
    grid_sync(bar);
    lstm_update(gates, c0, h0, (bf16*)nullptr, tid, nthr);
    grid_sync(bar);
    // ---- layer 1: gates = h0 @ Wih1^T + h1 @ Whh1^T + b1 ----
#pragma unroll
    for (int i = 0; i < 8; ++i) acc[i] = bv1;
#pragma unroll 4
    for (int k = 0; k < H_; k += 32)
      acc = wmma_bf16(load_a_frag(a_h0 + k), load_b_frag(b_i1 + k), acc);
#pragma unroll
    for (int k = 0; k < 16; ++k)
      acc = wmma_bf16(load_a_frag(a_h1 + 32 * k), bwh1[k], acc);
    store_tile(gates, H4_, mt, nt, lane, acc);
    grid_sync(bar);
    lstm_update(gates, c1, h1, (bf16*)nullptr, tid, nthr);
    grid_sync(bar);
    // ---- fc: y = h1 @ fcW^T + fcb ; write d_out and next xt ----
    if (do_fc) {
      v8f y;
#pragma unroll
      for (int i = 0; i < 8; ++i) y[i] = bvf;
#pragma unroll 4
      for (int k = 0; k < H_; k += 32)
        y = wmma_bf16(load_a_frag(a_fc + k), load_b_frag(b_fc + k), y);
      int r0 = fmt * 16 + ((lane >> 4) << 3);
      if (fcol < EIN_) {
#pragma unroll
        for (int i = 0; i < 8; ++i)
          out[(size_t)(r0 + i) * TP_ * EIN_ + (size_t)t * EIN_ + fcol] = y[i];
      }
      // fc rows >= 321 are zero-padded weights + zero bias -> y==0 keeps xt pad zero
#pragma unroll
      for (int i = 0; i < 8; ++i)
        xt[(size_t)(r0 + i) * KIN_ + fcol] = (bf16)y[i];
    }
    grid_sync(bar);  // xt visible to all before next step
  }
}

// ---------------- host launcher ----------------
extern "C" void kernel_launch(void* const* d_in, const int* in_sizes, int n_in,
                              void* d_out, int out_size, void* d_ws, size_t ws_size,
                              hipStream_t stream) {
  (void)in_sizes; (void)n_in; (void)out_size; (void)ws_size;
  const float* x     = (const float*)d_in[0];
  const float* eWih0 = (const float*)d_in[1];
  const float* eWhh0 = (const float*)d_in[2];
  const float* eb0   = (const float*)d_in[3];
  const float* eWih1 = (const float*)d_in[4];
  const float* eWhh1 = (const float*)d_in[5];
  const float* eb1   = (const float*)d_in[6];
  const float* dWih0 = (const float*)d_in[7];
  const float* dWhh0 = (const float*)d_in[8];
  const float* db0   = (const float*)d_in[9];
  const float* dWih1 = (const float*)d_in[10];
  const float* dWhh1 = (const float*)d_in[11];
  const float* db1   = (const float*)d_in[12];
  const float* fcW   = (const float*)d_in[13];
  const float* fcb   = (const float*)d_in[14];
  // d_in[15] = pred_len (device scalar) -- reference constant 96 hardcoded as TP_.

  char* ws = (char*)d_ws;
  size_t off = 0;
  auto take = [&](size_t bytes) -> char* {
    off = (off + 255) & ~(size_t)255;
    char* p = ws + off;
    off += bytes;
    return p;
  };
  unsigned* bar   = (unsigned*)take(256);
  bf16* wih0e_b = (bf16*)take((size_t)H4_ * KIN_ * 2);
  bf16* whh0e_b = (bf16*)take((size_t)H4_ * H_ * 2);
  bf16* wih1e_b = (bf16*)take((size_t)H4_ * H_ * 2);
  bf16* whh1e_b = (bf16*)take((size_t)H4_ * H_ * 2);
  bf16* wih0d_b = (bf16*)take((size_t)H4_ * KIN_ * 2);
  bf16* whh0d_b = (bf16*)take((size_t)H4_ * H_ * 2);
  bf16* wih1d_b = (bf16*)take((size_t)H4_ * H_ * 2);
  bf16* whh1d_b = (bf16*)take((size_t)H4_ * H_ * 2);
  bf16* fcw_b   = (bf16*)take((size_t)FCR_ * H_ * 2);
  bf16* x_bf    = (bf16*)take((size_t)T_ * B_ * KIN_ * 2);
  bf16* out0    = (bf16*)take((size_t)T_ * B_ * H_ * 2);
  float* xproj  = (float*)take((size_t)T_ * B_ * H4_ * 4);  // reused for both layers
  float* gates  = (float*)take((size_t)B_ * H4_ * 4);
  bf16* he0     = (bf16*)take((size_t)B_ * H_ * 2);
  bf16* he1     = (bf16*)take((size_t)B_ * H_ * 2);
  bf16* xt      = (bf16*)take((size_t)B_ * KIN_ * 2);
  float* ce0    = (float*)take((size_t)B_ * H_ * 4);
  float* ce1    = (float*)take((size_t)B_ * H_ * 4);
  // total workspace requirement ~230 MB

  PrepArgs pa;
  pa.x = x;
  pa.wih0e = eWih0; pa.whh0e = eWhh0; pa.wih1e = eWih1; pa.whh1e = eWhh1;
  pa.wih0d = dWih0; pa.whh0d = dWhh0; pa.wih1d = dWih1; pa.whh1d = dWhh1;
  pa.fcw = fcW;
  pa.x_bf = x_bf;
  pa.wih0e_b = wih0e_b; pa.whh0e_b = whh0e_b; pa.wih1e_b = wih1e_b; pa.whh1e_b = whh1e_b;
  pa.wih0d_b = wih0d_b; pa.whh0d_b = whh0d_b; pa.wih1d_b = wih1d_b; pa.whh1d_b = whh1d_b;
  pa.fcw_b = fcw_b;
  pa.he0 = he0; pa.he1 = he1; pa.xt = xt;
  pa.ce0 = ce0; pa.ce1 = ce1;
  pa.bar = bar;
  hipLaunchKernelGGL(prep_kernel, dim3(1024), dim3(256), 0, stream, pa);

  const int MT = (T_ * B_) / 16;  // 1344 M-tiles
  // encoder layer 0 input projection: xproj = x_bf @ Wih0^T + b0
  hipLaunchKernelGGL(gemm_bias_bf16, dim3(H4_ / 16 / 8, MT), dim3(256), 0, stream,
                     x_bf, KIN_, wih0e_b, KIN_, eb0, xproj, H4_, KIN_);
  // encoder layer 0 recurrence -> out0 (bf16 seq), final he0/ce0
  hipLaunchKernelGGL(lstm_recurrence, dim3(64), dim3(256), 0, stream,
                     whh0e_b, xproj, he0, ce0, out0, gates, bar, T_);
  // encoder layer 1 input projection: xproj = out0 @ Wih1^T + b1
  hipLaunchKernelGGL(gemm_bias_bf16, dim3(H4_ / 16 / 8, MT), dim3(256), 0, stream,
                     out0, H_, wih1e_b, H_, eb1, xproj, H4_, H_);
  // encoder layer 1 recurrence -> final he1/ce1
  hipLaunchKernelGGL(lstm_recurrence, dim3(64), dim3(256), 0, stream,
                     whh1e_b, xproj, he1, ce1, (bf16*)nullptr, gates, bar, T_);
  // autoregressive decoder: states continue from encoder finals
  hipLaunchKernelGGL(decoder_kernel, dim3(64), dim3(256), 0, stream,
                     wih0d_b, whh0d_b, db0, wih1d_b, whh1d_b, db1, fcw_b, fcb,
                     xt, he0, ce0, he1, ce1, gates, (float*)d_out, bar);
}
